// MultiBoxLoss_58050777973422
// MI455X (gfx1250) — compile-verified, matrix operand order checked by reading, not run on hardware
//
#include <hip/hip_runtime.h>
#include <hip/hip_bf16.h>

typedef __attribute__((ext_vector_type(16))) _Float16 v16h;
typedef __attribute__((ext_vector_type(8)))  float    v8f;

constexpr int BB = 16, PP = 19248, CC = 81, OO = 32;
constexpr int NBINS = 2048;
constexpr float BIN_SCALE = (float)NBINS / 16.0f;   // loss_c range [0,16) -> bins
constexpr size_t NBP = (size_t)BB * PP;             // 307968 = 1203*256 exactly

// ---- workspace layout (bytes) ----
constexpr size_t OFF_BTOV = 0;                       // float[NBP]  best_truth_overlap
constexpr size_t OFF_BTIX = OFF_BTOV + NBP * 4;      // int[NBP]    best_truth_idx
constexpr size_t OFF_CONF = OFF_BTIX + NBP * 4;      // int[NBP]    conf_t
constexpr size_t OFF_LC   = OFF_CONF + NBP * 4;      // float[NBP]  loss_c
constexpr size_t OFF_ZERO = OFF_LC   + NBP * 4;      // ---- zero-init region ----
constexpr size_t OFF_HIST = OFF_ZERO;                // int[BB*NBINS]
constexpr size_t OFF_BP   = OFF_HIST + (size_t)BB * NBINS * 4;   // u64[BB*OO] packed best-prior
constexpr size_t OFF_NPOS = OFF_BP   + (size_t)BB * OO * 8;      // int[BB]
constexpr size_t OFF_BCNT = OFF_NPOS + BB * 4;       // int[BB] boundary counters
constexpr size_t OFF_TPOS = OFF_BCNT + BB * 4;       // int[1]
constexpr size_t OFF_LB   = OFF_TPOS + 4;            // float loss_B acc
constexpr size_t OFF_CEP  = OFF_LB   + 4;            // float ce(pos) acc
constexpr size_t OFF_LCN  = OFF_CEP  + 4;            // float top-K neg acc
constexpr size_t OFF_ZEND = OFF_LCN  + 4;
constexpr size_t OFF_THR  = OFF_ZEND;                // int[BB] threshold bin
constexpr size_t OFF_NEED = OFF_THR + BB * 4;        // int[BB] count at boundary bin
constexpr int NZERO_INTS = (int)((OFF_ZEND - OFF_ZERO) / 4);

// ---------------- K0: zero the counter/hist region ----------------
__global__ __launch_bounds__(256) void k_init(int* z) {
    int i = blockIdx.x * 256 + threadIdx.x;
    if (i < NZERO_INTS) z[i] = 0;
}

// ---------------- K1: IoU, per-prior best GT, per-GT best prior ----------------
__global__ __launch_bounds__(256) void k_overlap(const float* __restrict__ gt_boxes,
                                                 const float* __restrict__ priors,
                                                 float* __restrict__ bt_ov,
                                                 int*   __restrict__ bt_idx,
                                                 unsigned long long* __restrict__ best_prior) {
    const int b = blockIdx.y;
    const int p = blockIdx.x * 256 + threadIdx.x;
    const int lane = threadIdx.x & 31;

    __shared__ float4 sgt[OO];
    if (threadIdx.x < OO)
        sgt[threadIdx.x] = ((const float4*)(gt_boxes + (size_t)b * OO * 4))[threadIdx.x];
    __syncthreads();

    const bool valid = p < PP;
    float px1 = 0.f, py1 = 0.f, px2 = 0.f, py2 = 0.f, areaP = 0.f;
    if (valid) {
        float4 pr = ((const float4*)priors)[p];          // cx, cy, w, h
        px1 = pr.x - 0.5f * pr.z;  py1 = pr.y - 0.5f * pr.w;
        px2 = pr.x + 0.5f * pr.z;  py2 = pr.y + 0.5f * pr.w;
        areaP = pr.z * pr.w;
    }
    float best = -1.0f; int bi = 0;
    for (int o = 0; o < OO; ++o) {
        float4 g = sgt[o];
        float iw = fminf(px2, g.z) - fmaxf(px1, g.x);
        float ih = fminf(py2, g.w) - fmaxf(py1, g.y);
        float inter = fmaxf(iw, 0.f) * fmaxf(ih, 0.f);
        float areaG = (g.z - g.x) * (g.w - g.y);
        float iou = inter / (areaP + areaG - inter);
        if (!valid) iou = -1.0f;
        if (iou > best) { best = iou; bi = o; }
        // per-GT argmax over priors: pack (iou bits | prior idx), wave-reduce, one atomic
        unsigned long long pk = valid
            ? ((unsigned long long)__float_as_uint(fmaxf(iou, 0.f)) << 32) | (unsigned)p
            : 0ull;
        #pragma unroll
        for (int off = 16; off > 0; off >>= 1) {
            unsigned long long o2 = __shfl_xor(pk, off, 32);
            pk = (o2 > pk) ? o2 : pk;
        }
        if (lane == 0) atomicMax(&best_prior[b * OO + o], pk);
    }
    if (valid) { bt_ov[(size_t)b * PP + p] = best; bt_idx[(size_t)b * PP + p] = bi; }
}

// ---------------- K2: GT override (serial per batch = ref semantics) ----------------
__global__ void k_override(const unsigned long long* __restrict__ best_prior,
                           float* __restrict__ bt_ov, int* __restrict__ bt_idx) {
    int b = threadIdx.x;
    if (b >= BB) return;
    for (int o = 0; o < OO; ++o) {
        int j = (int)(best_prior[b * OO + o] & 0xffffffffu);
        bt_ov[(size_t)b * PP + j] = 2.0f;
        bt_idx[(size_t)b * PP + j] = o;
    }
}

// ---------------- K3: match -> conf_t, encode, smooth-L1 loc loss ----------------
__global__ __launch_bounds__(256) void k_match(const float* __restrict__ bt_ov,
                                               const int* __restrict__ bt_idx,
                                               const int* __restrict__ gt_labels,
                                               const float* __restrict__ gt_boxes,
                                               const float* __restrict__ priors,
                                               const float* __restrict__ loc_data,
                                               int* __restrict__ conf_t,
                                               int* __restrict__ num_pos, int* __restrict__ total_pos,
                                               float* __restrict__ loss_B) {
    const size_t i = (size_t)blockIdx.x * 256 + threadIdx.x;   // exactly NBP threads
    const int b = (int)(i / PP);
    const int p = (int)(i % PP);
    const int lane = threadIdx.x & 31;

    float ov = bt_ov[i];
    int   o  = bt_idx[i];
    int ct = gt_labels[b * OO + o] + 1;
    if (ov < 0.5f) ct = -1;
    if (ov < 0.4f) ct = 0;
    conf_t[i] = ct;
    bool pos = ct > 0;

    float ll = 0.f;
    if (pos) {
        float4 g  = ((const float4*)(gt_boxes + (size_t)b * OO * 4))[o];  // x1,y1,x2,y2
        float4 pr = ((const float4*)priors)[p];                           // cx,cy,w,h
        float tcx = ((g.x + g.z) * 0.5f - pr.x) / (0.1f * pr.z);
        float tcy = ((g.y + g.w) * 0.5f - pr.y) / (0.1f * pr.w);
        float tw  = __logf((g.z - g.x) / pr.z) / 0.2f;
        float th  = __logf((g.w - g.y) / pr.w) / 0.2f;
        float4 ld = ((const float4*)loc_data)[i];
        float t[4] = {tcx, tcy, tw, th};
        float l[4] = {ld.x, ld.y, ld.z, ld.w};
        #pragma unroll
        for (int r = 0; r < 4; ++r) {
            float d = fabsf(l[r] - t[r]);
            ll += (d < 1.0f) ? 0.5f * d * d : d - 0.5f;
        }
        atomicAdd(&num_pos[b], 1);     // waves can straddle batches; per-pos atomic is cheap
    }
    unsigned long long bal = __ballot(pos);
    #pragma unroll
    for (int off = 16; off > 0; off >>= 1) ll += __shfl_down(ll, off, 32);
    if (lane == 0) {
        if (ll != 0.f) atomicAdd(loss_B, ll);
        int np = __popcll(bal);
        if (np) atomicAdd(total_pos, np);
    }
}

// ---------------- K4: logsumexp; async global->LDS staging + WMMA class-sum ----------------
// One wave per 16-prior tile. The tile (16 rows x 81 classes) is 5184 contiguous,
// 16B-aligned bytes: stage it with GLOBAL_LOAD_ASYNC_TO_LDS_B128 (ASYNCcnt), then do
// both the max pass and the exp pass from LDS -> a single HBM read of conf_data.
constexpr int TILE_F = CC * 16;          // 1296 floats = 5184 B = 324 float4
constexpr int TILE_V4 = TILE_F / 4;      // 324

__global__ __launch_bounds__(256) void k_conf(const float* __restrict__ conf,
                                              const int* __restrict__ conf_t,
                                              float* __restrict__ loss_c,
                                              int* __restrict__ hist,
                                              float* __restrict__ ce_pos) {
    const int wave = threadIdx.x >> 5;
    const int lane = threadIdx.x & 31;
    const int tile = blockIdx.x * 8 + wave;        // [0, 19248): full waves only
    const int b  = tile / (PP / 16);
    const int p0 = (tile % (PP / 16)) * 16;
    const int half = lane >> 4;
    const int m = lane & 15;

    __shared__ float smem[8 * TILE_F];             // 41472 B per block
    float* tilebuf = smem + wave * TILE_F;

    // ---- async stage: 324 coalesced b128 transfers into this wave's LDS slice ----
    const char* gbase = (const char*)(conf + (size_t)(b * PP + p0) * CC);
    unsigned lds_base = (unsigned)(size_t)tilebuf;           // flat addr low 32b = LDS offset
    #pragma unroll
    for (int it = 0; it < (TILE_V4 + 31) / 32; ++it) {
        int idx = it * 32 + lane;
        if (idx < TILE_V4) {
            unsigned long long ga = (unsigned long long)(size_t)gbase + (unsigned)idx * 16u;
            unsigned la = lds_base + (unsigned)idx * 16u;
            asm volatile("global_load_async_to_lds_b128 %0, %1, off"
                         :: "v"(la), "v"(ga) : "memory");
        }
    }
    asm volatile("s_wait_asynccnt 0x0" ::: "memory");

    const float* row = tilebuf + m * CC;

    // pass 1: per-prior max (split across lane pair m / m+16), from LDS
    float mx = -1e30f;
    #pragma unroll
    for (int c = 0; c < 3; ++c)
        #pragma unroll
        for (int e = 0; e < 16; ++e) {
            int k = 32 * c + e + 8 * ((e >> 3) + half);
            if (k < CC) mx = fmaxf(mx, row[k]);
        }
    mx = fmaxf(mx, __shfl_xor(mx, 16, 32));

    // pass 2: exp -> f16 A-tiles; B = ones => D rows are class sums
    v16h ones;
    #pragma unroll
    for (int e = 0; e < 16; ++e) ones[e] = (_Float16)1.0f;
    v8f acc = {};
    #pragma unroll
    for (int c = 0; c < 3; ++c) {
        v16h a;
        #pragma unroll
        for (int e = 0; e < 16; ++e) {
            int k = 32 * c + e + 8 * ((e >> 3) + half);
            float v = (k < CC) ? __expf(row[k] - mx) : 0.0f;
            a[e] = (_Float16)v;
        }
        acc = __builtin_amdgcn_wmma_f32_16x16x32_f16(false, a, false, ones,
                                                     (short)0, acc, false, false);
    }

    // writers: lanes 0..7 -> rows 0..7 (acc[lane]); lanes 24..31 -> rows 8..15 (acc[lane-24])
    bool writer = (lane < 8) || (lane >= 24);
    int  j = (lane < 8) ? lane : lane - 24;
    float s = 0.f;
    #pragma unroll
    for (int r = 0; r < 8; ++r) if (r == j) s = acc[r];
    if (writer) {
        int mm = (lane < 8) ? lane : lane - 16;          // == this lane's mx owner (lane&15)
        size_t idx = (size_t)b * PP + p0 + mm;
        const float* rw = tilebuf + mm * CC;             // x0 / x_tgt straight from LDS
        float logZ = mx + __logf(s);
        int ct = conf_t[idx];
        float lc = 0.f;
        if (ct > 0) {
            atomicAdd(ce_pos, logZ - rw[ct]);
        } else if (ct == 0) {
            lc = logZ - rw[0];
            int bin = (int)(lc * BIN_SCALE);
            bin = bin < 0 ? 0 : (bin > NBINS - 1 ? NBINS - 1 : bin);
            atomicAdd(&hist[b * NBINS + bin], 1);
        }
        loss_c[idx] = lc;
    }
}

// ---------------- K6: per-batch histogram threshold scan ----------------
__global__ void k_thresh(const int* __restrict__ hist, const int* __restrict__ num_pos,
                         int* __restrict__ thr, int* __restrict__ need) {
    int b = threadIdx.x;
    if (b >= BB) return;
    int K = num_pos[b] * 3;
    if (K > PP - 1) K = PP - 1;
    int t = NBINS, nd = 0;
    if (K > 0) {
        int cum = 0; t = -1;                      // t=-1 => take every eligible negative
        for (int bin = NBINS - 1; bin >= 0; --bin) {
            int h = hist[b * NBINS + bin];
            if (cum + h >= K) { t = bin; nd = K - cum; break; }
            cum += h;
        }
    }
    thr[b] = t; need[b] = nd;
}

// ---------------- K7: sum top-K negative losses ----------------
__global__ __launch_bounds__(256) void k_negsum(const int* __restrict__ conf_t,
                                                const float* __restrict__ loss_c,
                                                const int* __restrict__ thr,
                                                const int* __restrict__ need,
                                                int* __restrict__ bcount,
                                                float* __restrict__ loss_C_neg) {
    const size_t i = (size_t)blockIdx.x * 256 + threadIdx.x;
    const int b = (int)(i / PP);
    float contrib = 0.f;
    if (conf_t[i] == 0) {
        float lc = loss_c[i];
        int bin = (int)(lc * BIN_SCALE);
        bin = bin < 0 ? 0 : (bin > NBINS - 1 ? NBINS - 1 : bin);
        int t = thr[b];
        if (bin > t) contrib = lc;
        else if (bin == t) {
            int n = atomicAdd(&bcount[b], 1);
            if (n < need[b]) contrib = lc;
        }
    }
    __shared__ float red[256];
    red[threadIdx.x] = contrib;
    __syncthreads();
    for (int s = 128; s > 0; s >>= 1) {
        if (threadIdx.x < s) red[threadIdx.x] += red[threadIdx.x + s];
        __syncthreads();
    }
    if (threadIdx.x == 0 && red[0] != 0.f) atomicAdd(loss_C_neg, red[0]);
}

// ---------------- K8: finalize ----------------
__global__ void k_final(const float* __restrict__ loss_B, const float* __restrict__ ce_pos,
                        const float* __restrict__ loss_C_neg, const int* __restrict__ total_pos,
                        float* __restrict__ out) {
    float tp = (float)*total_pos;
    out[0] = 1.5f * (*loss_B) / tp;
    out[1] = (*ce_pos + *loss_C_neg) / tp;
}

extern "C" void kernel_launch(void* const* d_in, const int* in_sizes, int n_in,
                              void* d_out, int out_size, void* d_ws, size_t ws_size,
                              hipStream_t stream) {
    const float* loc_data  = (const float*)d_in[0];
    const float* conf_data = (const float*)d_in[1];
    const float* priors    = (const float*)d_in[2];
    const float* gt_boxes  = (const float*)d_in[3];
    const int*   gt_labels = (const int*)d_in[4];
    float* out = (float*)d_out;

    char* ws = (char*)d_ws;
    float* bt_ov   = (float*)(ws + OFF_BTOV);
    int*   bt_idx  = (int*)  (ws + OFF_BTIX);
    int*   conf_t  = (int*)  (ws + OFF_CONF);
    float* loss_c  = (float*)(ws + OFF_LC);
    int*   hist    = (int*)  (ws + OFF_HIST);
    unsigned long long* best_prior = (unsigned long long*)(ws + OFF_BP);
    int*   num_pos = (int*)  (ws + OFF_NPOS);
    int*   bcount  = (int*)  (ws + OFF_BCNT);
    int*   tpos    = (int*)  (ws + OFF_TPOS);
    float* lossB   = (float*)(ws + OFF_LB);
    float* cepos   = (float*)(ws + OFF_CEP);
    float* lossCn  = (float*)(ws + OFF_LCN);
    int*   thr     = (int*)  (ws + OFF_THR);
    int*   need    = (int*)  (ws + OFF_NEED);

    k_init<<<(NZERO_INTS + 255) / 256, 256, 0, stream>>>((int*)(ws + OFF_ZERO));
    k_overlap<<<dim3((PP + 255) / 256, BB), 256, 0, stream>>>(gt_boxes, priors, bt_ov, bt_idx, best_prior);
    k_override<<<1, 32, 0, stream>>>(best_prior, bt_ov, bt_idx);
    k_match<<<(int)(NBP / 256), 256, 0, stream>>>(bt_ov, bt_idx, gt_labels, gt_boxes, priors,
                                                  loc_data, conf_t, num_pos, tpos, lossB);
    k_conf<<<(BB * (PP / 16)) / 8, 256, 0, stream>>>(conf_data, conf_t, loss_c, hist, cepos);
    k_thresh<<<1, 32, 0, stream>>>(hist, num_pos, thr, need);
    k_negsum<<<(int)(NBP / 256), 256, 0, stream>>>(conf_t, loss_c, thr, need, bcount, lossCn);
    k_final<<<1, 1, 0, stream>>>(lossB, cepos, lossCn, tpos, out);
}
